// Barrier_Net_52553219833862
// MI455X (gfx1250) — matrix operand order repaired and checked by reading, not compiled, not run on hardware
//
#include <hip/hip_runtime.h>
#include <math.h>

typedef __attribute__((ext_vector_type(16))) _Float16 v16h;
typedef __attribute__((ext_vector_type(8)))  _Float16 v8h;
typedef __attribute__((ext_vector_type(2)))  _Float16 v2h;
typedef __attribute__((ext_vector_type(8)))  float    v8f;

namespace {

constexpr int   kXW = 53;                 // 1 + 9*4 + 16
constexpr int   kNN = 8, kNO = 8, kSD = 4;
constexpr float kRAgent = 0.2f, kBGamma = 0.05f, kDeltaR = 0.5f;
constexpr float kPhiMax = 0.5f, kAMax = 1.0f;
constexpr int   kSST    = 72;             // staging row stride (f16 elems); 144B, 16B-multiple
constexpr int   kWaves  = 4;              // 16-sample tiles per block
constexpr int   kWsStride = 6;

// 13 weight matrices: phiN0..2, rhoN0..1, phiO0..2, rhoO0..1, psi0..2
constexpr int kNW = 13;
constexpr int wKd[kNW]   = {4,64,64, 16,64, 2,64,64, 16,64, 36,64,64};
constexpr int wNd[kNW]   = {64,64,16, 64,16, 64,64,16, 64,16, 64,64,2};
constexpr int wNT[kNW]   = {4,4,1, 4,1, 4,4,1, 4,1, 4,4,1};   // ceil(N/16)
constexpr int wKT[kNW]   = {1,2,2, 1,2, 1,2,2, 1,2, 2,2,2};   // ceil(K/32)
constexpr int wSw[kNW]   = {0,1,1, 1,1, 0,1,1, 1,1, 1,1,1};   // pre-swizzled?
constexpr int wTOff[kNW] = {0,0,8, 10,14, 0,16,24, 26,30, 32,40,48};  // tile offset (1 tile = 512 f16)
constexpr int wCOff[kNW] = {0,0,0, 0,0, 256,0,0, 0,0, 0,0,0};         // compact offset
constexpr int bOff[kNW]  = {0,64,128, 144,208, 224,288,352, 368,432, 448,512,576};
constexpr int kSwTiles = 50;              // 50 KB of swizzled f16 fragments
constexpr int kCompact = 384;             // phiN0 (256) + phiO0 (128)
constexpr int kBTot    = 578;

__device__ __forceinline__ v8f wmma_f16(v16h a, v16h b, v8f c) {
  // (neg_a, A, neg_b, B, c_mod, C, reuse_a, reuse_b)
  return __builtin_amdgcn_wmma_f32_16x16x32_f16(false, a, false, b, (short)0, c, false, false);
}

// B fragment: pre-swizzled, 32B contiguous per lane -> 2x ds_load_b128
__device__ __forceinline__ v16h load_b_sw(const _Float16* tiles, int t, int lane) {
  const _Float16* p = tiles + (t * 32 + lane) * 16;
  v8h lo = *(const v8h*)p;
  v8h hi = *(const v8h*)(p + 8);
  return __builtin_shufflevector(lo, hi, 0,1,2,3,4,5,6,7,8,9,10,11,12,13,14,15);
}

// A fragment (16x32 MxK) from f16 row-major staging:
// lane%16 = row m; element i -> K = kt*32 + g*8 + (i&7) + (i>=8)*16
// => two contiguous 16B runs -> 2x ds_load_b128
__device__ __forceinline__ v16h load_a_frag(const _Float16* Sh, int kt, int lane) {
  const int m = lane & 15, g = lane >> 4;
  const _Float16* p = Sh + m * kSST + kt * 32 + g * 8;
  v8h lo = *(const v8h*)p;
  v8h hi = *(const v8h*)(p + 16);
  return __builtin_shufflevector(lo, hi, 0,1,2,3,4,5,6,7,8,9,10,11,12,13,14,15);
}

// write a C/D 16x16 f32 tile into f16 staging at column offset, bias*scale added
__device__ __forceinline__ void stage_reg16_h(_Float16* Sh, v8f c, const float* bias,
                                              float bscale, int colOff, int lane) {
  const int n = lane & 15, g = lane >> 4;
  const float b = bscale * bias[n];
#pragma unroll
  for (int r = 0; r < 8; ++r)
    Sh[(g * 8 + r) * kSST + colOff + n] = (_Float16)(c[r] + b);
}

__device__ __forceinline__ void zero_cols_h(_Float16* Sh, int c0, int c1, int lane) {
  const int wp = (c1 - c0) >> 1;
  for (int idx = lane; idx < 16 * wp; idx += 32) {
    const int r = idx / wp, p = idx % wp;
    *(unsigned*)(Sh + r * kSST + c0 + 2 * p) = 0u;
  }
}

// stage x features [featOff, featOff+width) into cols [colBase, colBase+32), zero-padded
__device__ __forceinline__ void stage_x(_Float16* Sh, const float* x, int s0,
                                        int colBase, int featOff, int width, int lane) {
  for (int idx = lane; idx < 16 * 16; idx += 32) {
    const int r = idx >> 4, c = (idx & 15) * 2;
    const float* xr = x + (long)(s0 + r) * kXW + featOff;
    v2h v;
    v[0] = (c     < width) ? (_Float16)xr[c]     : (_Float16)0.0f;
    v[1] = (c + 1 < width) ? (_Float16)xr[c + 1] : (_Float16)0.0f;
    *(v2h*)(Sh + r * kSST + colBase + c) = v;
  }
}

// dense layer, pre-swizzled weights, staging in/out in place (per-wave buffer:
// LDS ops from one wave are in order -> no block barrier needed).
// All B fragments prefetched so the WMMA chain runs without per-tile dscnt stalls.
template <int KT, int NT>
__device__ __forceinline__ void dense_sw(_Float16* Sh, const _Float16* tiles,
                                         const float* bias, int Nout, bool relu,
                                         int lane) {
  v16h a[KT];
#pragma unroll
  for (int t = 0; t < KT; ++t) a[t] = load_a_frag(Sh, t, lane);
  v16h b[KT * NT];
#pragma unroll
  for (int u = 0; u < KT * NT; ++u) b[u] = load_b_sw(tiles, u, lane);
  const int nb = lane & 15, g = lane >> 4;
#pragma unroll
  for (int nt = 0; nt < NT; ++nt) {
    v8f c = {};
#pragma unroll
    for (int t = 0; t < KT; ++t) c = wmma_f16(a[t], b[t * NT + nt], c);
    const int n = nt * 16 + nb;
    const bool valid = n < Nout;
    const float bb = valid ? bias[n] : 0.0f;
#pragma unroll
    for (int r = 0; r < 8; ++r) {
      float v = valid ? c[r] + bb : 0.0f;
      if (relu) v = fmaxf(v, 0.0f);
      Sh[(g * 8 + r) * kSST + n] = (_Float16)v;
    }
  }
}

// first layer (tiny K): compact [K][64] f16 weights, guarded fragment build
__device__ __forceinline__ void dense_l1(_Float16* Sh, const _Float16* Wc, int Kdim,
                                         const float* bias, int lane) {
  v16h a = load_a_frag(Sh, 0, lane);
  const int nb = lane & 15;
  const int g  = lane >> 4;
#pragma unroll
  for (int nt = 0; nt < 4; ++nt) {
    const int n = nt * 16 + nb;
    v16h b;
#pragma unroll
    for (int i = 0; i < 16; ++i)
      b[i] = (g == 0 && i < Kdim) ? Wc[i * 64 + n] : (_Float16)0.0f;
    v8f c = {};
    c = wmma_f16(a, b, c);
#pragma unroll
    for (int r = 0; r < 8; ++r)
      Sh[(g * 8 + r) * kSST + n] = (_Float16)fmaxf(c[r] + bias[n], 0.0f);
  }
}

// final MLP layer (N=16, NT==1): accumulate straight into a register tile
template <int KT>
__device__ __forceinline__ v8f dense_to_reg(const _Float16* Sh, const _Float16* tiles,
                                            v8f acc, int lane) {
  v16h a[KT], b[KT];
#pragma unroll
  for (int t = 0; t < KT; ++t) { a[t] = load_a_frag(Sh, t, lane); b[t] = load_b_sw(tiles, t, lane); }
#pragma unroll
  for (int t = 0; t < KT; ++t) acc = wmma_f16(a[t], b[t], acc);
  return acc;
}

struct BNParams {
  const float* x;
  const float* W[kNW];
  const float* B[kNW];
  float* ws;
  int nd;
};

}  // namespace

__global__ void __launch_bounds__(kWaves * 32)
bnet_forward(BNParams P) {
  __shared__ __align__(16) _Float16 sWs[kSwTiles * 512];
  __shared__ __align__(16) _Float16 sWc[kCompact];
  __shared__ float sB[kBTot];
  __shared__ __align__(16) _Float16 sStage[kWaves * 16 * kSST];

  const int tid  = threadIdx.x;
  const int lane = tid & 31;
  const int wave = tid >> 5;

  // Stage weights into LDS: big matrices pre-swizzled into WMMA B-fragment order,
  // tiny layer-1 matrices compact; biases in f32.
  for (int w = 0; w < kNW; ++w) {
    const float* src = P.W[w];
    if (wSw[w]) {
      _Float16* dst = sWs + wTOff[w] * 512;
      const int K = wKd[w], N = wNd[w], NT = wNT[w];
      const int tot = wKT[w] * NT * 512;
      for (int idx = tid; idx < tot; idx += blockDim.x) {
        const int t  = idx >> 9;          // fragment tile
        const int ln = (idx >> 4) & 31;   // lane slot
        const int i  = idx & 15;          // element within lane
        const int kt = t / NT, nt = t - kt * NT;
        const int n  = nt * 16 + (ln & 15);
        const int k  = kt * 32 + (ln >> 4) * 8 + (i & 7) + (i >> 3) * 16;
        dst[idx] = (k < K && n < N) ? (_Float16)src[k * N + n] : (_Float16)0.0f;
      }
    } else {
      _Float16* dst = sWc + wCOff[w];
      const int cnt = wKd[w] * wNd[w];
      for (int idx = tid; idx < cnt; idx += blockDim.x) dst[idx] = (_Float16)src[idx];
    }
    const float* bs = P.B[w];
    for (int i = tid; i < wNd[w]; i += blockDim.x) sB[bOff[w] + i] = bs[i];
  }
  __syncthreads();   // the only cross-wave dependency in the kernel

  int s0 = (blockIdx.x * kWaves + wave) * 16;
  if (s0 + 16 > P.nd) s0 = P.nd - 16;   // clamp: duplicate work, identical result
  _Float16* Sh = sStage + wave * 16 * kSST;
  const float* x = P.x;

  // ---- DeepSet over neighbors: sum_j phi_n, summed in the WMMA C accumulator ----
  v8f accN = {};
  for (int j = 0; j < kNN; ++j) {
    stage_x(Sh, x, s0, 0, 1 + kSD + kSD * j, kSD, lane);
    dense_l1(Sh, sWc + wCOff[0], 4, sB + bOff[0], lane);
    dense_sw<2, 4>(Sh, sWs + wTOff[1] * 512, sB + bOff[1], 64, true, lane);
    accN = dense_to_reg<2>(Sh, sWs + wTOff[2] * 512, accN, lane);
  }
  stage_reg16_h(Sh, accN, sB + bOff[2], (float)kNN, 0, lane);  // + NN*b (summed bias)
  zero_cols_h(Sh, 16, 32, lane);
  dense_sw<1, 4>(Sh, sWs + wTOff[3] * 512, sB + bOff[3], 64, true, lane);
  v8f zacc = {};
  v8f rn = dense_to_reg<2>(Sh, sWs + wTOff[4] * 512, zacc, lane);

  // ---- DeepSet over obstacles ----
  v8f accO = {};
  for (int j = 0; j < kNO; ++j) {
    stage_x(Sh, x, s0, 0, 1 + kSD * (kNN + 1) + 2 * j, 2, lane);
    dense_l1(Sh, sWc + wCOff[5], 2, sB + bOff[5], lane);
    dense_sw<2, 4>(Sh, sWs + wTOff[6] * 512, sB + bOff[6], 64, true, lane);
    accO = dense_to_reg<2>(Sh, sWs + wTOff[7] * 512, accO, lane);
  }
  stage_reg16_h(Sh, accO, sB + bOff[7], (float)kNO, 0, lane);
  zero_cols_h(Sh, 16, 32, lane);
  dense_sw<1, 4>(Sh, sWs + wTOff[8] * 512, sB + bOff[8], 64, true, lane);
  v8f ro = dense_to_reg<2>(Sh, sWs + wTOff[9] * 512, zacc, lane);

  // ---- psi([rn, ro, g]) ----
  stage_reg16_h(Sh, rn, sB + bOff[4], 1.0f, 0, lane);   // rho last-layer bias folded in
  stage_reg16_h(Sh, ro, sB + bOff[9], 1.0f, 16, lane);
  stage_x(Sh, x, s0, 32, 1, kSD, lane);                 // cols 32..35 = g, 36..63 zero
  dense_sw<2, 4>(Sh, sWs + wTOff[10] * 512, sB + bOff[10], 64, true, lane);
  dense_sw<2, 4>(Sh, sWs + wTOff[11] * 512, sB + bOff[11], 64, true, lane);
  dense_sw<2, 1>(Sh, sWs + wTOff[12] * 512, sB + bOff[12], 2, false, lane);

  // ---- per-sample scalar tail: PHI_MAX clamp + barrier + minH ----
  if (lane < 16) {
    const int i = s0 + lane;
    const float* xr = x + (long)i * kXW;
    float ex = (float)Sh[lane * kSST + 0];
    float ey = (float)Sh[lane * kSST + 1];
    float inv = fmaxf(sqrtf(ex * ex + ey * ey) * (1.0f / kPhiMax), 1.0f);
    ex /= inv; ey /= inv;
    float bx = 0.0f, by = 0.0f, minH = 3.0e38f;
#pragma unroll
    for (int j = 0; j < kNN; ++j) {
      const float px = xr[1 + kSD + kSD * j], py = xr[1 + kSD + kSD * j + 1];
      const float np = sqrtf(px * px + py * py);
      const float H = np - 2.0f * kRAgent;
      minH = fminf(minH, H);
      const float d = 1.0f / (H * np);
      bx += px * d; by += py * d;
    }
#pragma unroll
    for (int j = 0; j < kNO; ++j) {
      const float pox = xr[1 + kSD * (kNN + 1) + 2 * j];
      const float poy = xr[1 + kSD * (kNN + 1) + 2 * j + 1];
      const float cx = fminf(fmaxf(0.0f, -pox - 0.5f), -pox + 0.5f);
      const float cy = fminf(fmaxf(0.0f, -poy - 0.5f), -poy + 0.5f);
      const float Ho = sqrtf(cx * cx + cy * cy) - kRAgent;
      minH = fminf(minH, Ho);
      const float npo = sqrtf(pox * pox + poy * poy);
      const float d = 1.0f / (Ho * npo);
      bx += pox * d; by += poy * d;
    }
    bx *= -kBGamma; by *= -kBGamma;
    float* w = P.ws + (long)i * kWsStride;
    w[0] = ex; w[1] = ey; w[2] = bx; w[3] = by; w[4] = minH;
  }
}

__global__ void bnet_finalize(const float* __restrict__ ws, float* __restrict__ out, int nd) {
  const int i = blockIdx.x * blockDim.x + threadIdx.x;
  if (i >= nd) return;
  // val comes from sample 0 (the reference's "[0]" quirk), broadcast to all
  const float e0x = ws[0], e0y = ws[1], b0x = ws[2], b0y = ws[3];
  const float val =
      fminf(sqrtf(b0x * b0x + b0y * b0y) / sqrtf(e0x * e0x + e0y * e0y), 1.0f);
  const float* w = ws + (long)i * kWsStride;
  const float s = (w[4] < kDeltaR) ? val : 1.0f;
  const float ax = s * w[0] + w[2];
  const float ay = s * w[1] + w[3];
  const float inv = fmaxf(sqrtf(ax * ax + ay * ay) * (1.0f / kAMax), 1.0f);
  out[2 * i + 0] = ax / inv;
  out[2 * i + 1] = ay / inv;
}

extern "C" void kernel_launch(void* const* d_in, const int* in_sizes, int n_in,
                              void* d_out, int out_size, void* d_ws, size_t ws_size,
                              hipStream_t stream) {
  (void)n_in; (void)out_size; (void)ws_size;
  BNParams P;
  P.x = (const float*)d_in[0];
  const int wi[kNW] = {1, 3, 5, 7, 9, 11, 13, 15, 17, 19, 21, 23, 25};
  for (int i = 0; i < kNW; ++i) {
    P.W[i] = (const float*)d_in[wi[i]];
    P.B[i] = (const float*)d_in[wi[i] + 1];
  }
  P.ws = (float*)d_ws;
  P.nd = in_sizes[0] / kXW;

  const int tiles  = (P.nd + 15) / 16;
  const int blocks = (tiles + kWaves - 1) / kWaves;
  bnet_forward<<<blocks, kWaves * 32, 0, stream>>>(P);
  bnet_finalize<<<(P.nd + 255) / 256, 256, 0, stream>>>(P.ws, (float*)d_out, P.nd);
}
